// SelfMHA_33492154974395
// MI455X (gfx1250) — matrix operand order repaired and checked by reading, not compiled, not run on hardware
//
#include <hip/hip_runtime.h>

typedef __bf16 bf16;
typedef __attribute__((ext_vector_type(16))) __bf16 v16bf;
typedef __attribute__((ext_vector_type(8)))  __bf16 v8bf;
typedef __attribute__((ext_vector_type(8)))  float  v8f;

#define B_  2
#define S_  2048
#define D_  1024
#define H_  16
#define DK_ 64

__device__ __forceinline__ v8f wmma_bf16(v16bf a, v16bf b, v8f c) {
  return __builtin_amdgcn_wmma_f32_16x16x32_bf16(false, a, false, b, (short)0, c,
                                                 false, false);
}

// A-matrix fragment (16x32, M x K), mat row-major [m][k] with leading dim ld.
// element i, lane L -> A[m0 + L%16][k0 + (i%8) + 8*(L/16) + 16*(i/8)]
__device__ __forceinline__ v16bf load_a_frag(const bf16* __restrict__ mat, int ld,
                                             int m0, int k0, int lane) {
  const bf16* p = mat + (size_t)(m0 + (lane & 15)) * ld + k0 + ((lane >> 4) << 3);
  v8bf lo = *(const v8bf*)(p);
  v8bf hi = *(const v8bf*)(p + 16);
  v16bf r;
#pragma unroll
  for (int i = 0; i < 8; ++i) { r[i] = lo[i]; r[i + 8] = hi[i]; }
  return r;
}

// B-matrix fragment (32x16, K x N) supplied as the TRANSPOSED matrix:
// WMMA B[kk][nn] = mat[n0 + nn][k0 + kk], mat row-major with leading dim ld.
// element i, lane L -> mat[n0 + L%16][k0 + (i%8) + 8*(L/16) + 16*(i/8)]
__device__ __forceinline__ v16bf load_bT_frag(const bf16* __restrict__ mat, int ld,
                                              int n0, int k0, int lane) {
  const bf16* p = mat + (size_t)(n0 + (lane & 15)) * ld + k0 + ((lane >> 4) << 3);
  v8bf lo = *(const v8bf*)(p);
  v8bf hi = *(const v8bf*)(p + 16);
  v16bf r;
#pragma unroll
  for (int i = 0; i < 8; ++i) { r[i] = lo[i]; r[i + 8] = hi[i]; }
  return r;
}

// ---------------------------------------------------------------- converts
__global__ void cvt_f32_bf16(const float* __restrict__ in, bf16* __restrict__ out,
                             int n) {
  int i = blockIdx.x * blockDim.x + threadIdx.x;
  if (i < n) out[i] = (bf16)in[i];
}

// ---------------------------------------------------------------- QKV GEMM
// qkv[m][e] = sum_d X[m][d] * W[e][d]; 32x64 tile per wave (8 WMMA / 6 fragment loads per
// K-chunk); routes into head-major Q/K and transposed V.
__global__ __launch_bounds__(128) void qkv_gemm(const bf16* __restrict__ X,
                                                const bf16* __restrict__ W,
                                                bf16* __restrict__ Qb,
                                                bf16* __restrict__ Kb,
                                                bf16* __restrict__ Vt) {
  const int lane = threadIdx.x & 31;
  const int wv   = threadIdx.x >> 5;
  const int row0 = blockIdx.x * 32;                 // 0..4064
  const int col0 = (blockIdx.y * 4 + wv) * 64;      // 0..3008
  const v8f vzero = {0.f, 0.f, 0.f, 0.f, 0.f, 0.f, 0.f, 0.f};
  v8f acc[8];
#pragma unroll
  for (int i = 0; i < 8; ++i) acc[i] = vzero;

  for (int kc = 0; kc < D_; kc += 32) {
    v16bf a0 = load_a_frag(X, D_, row0, kc, lane);
    v16bf a1 = load_a_frag(X, D_, row0 + 16, kc, lane);
#pragma unroll
    for (int g = 0; g < 4; ++g) {
      v16bf b = load_bT_frag(W, D_, col0 + g * 16, kc, lane);
      acc[2 * g]     = wmma_bf16(a0, b, acc[2 * g]);
      acc[2 * g + 1] = wmma_bf16(a1, b, acc[2 * g + 1]);
    }
    if (kc + 128 < D_) {   // prefetch next-next chunk lines (global_prefetch_b8)
      __builtin_prefetch(X + (size_t)(row0 + lane) * D_ + kc + 128, 0, 1);
      __builtin_prefetch(W + (size_t)(col0 + (lane & 31)) * D_ + kc + 128, 0, 1);
      __builtin_prefetch(W + (size_t)(col0 + 32 + (lane & 31)) * D_ + kc + 128, 0, 1);
    }
  }

#pragma unroll
  for (int g = 0; g < 4; ++g) {
    const int e   = col0 + g * 16 + (lane & 15);    // output channel
    const int seg = e >> 10;                        // 0=Q 1=K 2=V (uniform/wave)
    const int eh  = e & (D_ - 1);
    const int h   = eh & 15;
    const int d   = eh >> 4;
#pragma unroll
    for (int rblk = 0; rblk < 2; ++rblk) {
      const int rb = row0 + rblk * 16 + ((lane >> 4) << 3);
      const int bb = rb >> 11;                      // tile never crosses batch
      const int s0 = rb & (S_ - 1);
      const v8f a  = acc[2 * g + rblk];
      if (seg < 2) {
        bf16* dst = (seg == 0 ? Qb : Kb) +
                    (((size_t)(bb * H_ + h)) * S_ + s0) * DK_ + d;
#pragma unroll
        for (int r = 0; r < 8; ++r) dst[(size_t)r * DK_] = (bf16)a[r];
      } else {
        bf16* dst = Vt + (((size_t)(bb * H_ + h)) * DK_ + d) * S_ + s0;
#pragma unroll
        for (int r = 0; r < 8; ++r) dst[r] = (bf16)a[r];
      }
    }
  }
}

// ---------------------------------------------------------------- attention
// One 16-row q-tile per wave, 64-key chunks, flash-style online softmax.
__global__ __launch_bounds__(256) void attn_kernel(const bf16* __restrict__ Qb,
                                                   const bf16* __restrict__ Kb,
                                                   const bf16* __restrict__ Vt,
                                                   const unsigned char* __restrict__ mask,
                                                   bf16* __restrict__ atn) {
  __shared__ bf16 plds[8][16][64];            // per-wave P re-layout staging, 16 KB
  const int lane  = threadIdx.x & 31;
  const int wv    = threadIdx.x >> 5;
  const int t     = blockIdx.x * 8 + wv;      // 0..4095
  const int qTile = t & 127;
  const int bh    = t >> 7;                   // 0..31
  const int h     = bh & 15;
  const int bb    = bh >> 4;
  const bf16* Q = Qb + (size_t)bh * S_ * DK_;
  const bf16* K = Kb + (size_t)bh * S_ * DK_;
  const bf16* V = Vt + (size_t)bh * DK_ * S_;
  const unsigned char* M = mask + (size_t)bb * S_ * S_;
  const int q0 = qTile * 16;
  const int qrow_base = q0 + ((lane >> 4) << 3);

  v16bf qa0 = load_a_frag(Q, DK_, q0, 0, lane);
  v16bf qa1 = load_a_frag(Q, DK_, q0, 32, lane);

  float mrun[8], lrun[8];
  v8f acc0 = {}, acc1 = {}, acc2 = {}, acc3 = {};
#pragma unroll
  for (int r = 0; r < 8; ++r) { mrun[r] = -3.0e38f; lrun[r] = 0.0f; }
  const float scale = 0.125f;                 // 1/sqrt(64)

  for (int k0 = 0; k0 < S_; k0 += 64) {
    // scores: four 16x16 tiles over this 64-key chunk (2 WMMA each, K=dk=64)
    v8f s[4];
#pragma unroll
    for (int tt = 0; tt < 4; ++tt) {
      v8f sv = {};
      sv = wmma_bf16(qa0, load_bT_frag(K, DK_, k0 + 16 * tt, 0, lane), sv);
      sv = wmma_bf16(qa1, load_bT_frag(K, DK_, k0 + 16 * tt, 32, lane), sv);
      s[tt] = sv;
    }
    if (k0 + 64 < S_) {    // prefetch next chunk of K and V (global_prefetch_b8)
      __builtin_prefetch(K + (size_t)(k0 + 64 + lane) * DK_, 0, 1);
      __builtin_prefetch(K + (size_t)(k0 + 96 + lane) * DK_, 0, 1);
      __builtin_prefetch(V + (size_t)lane * S_ + k0 + 64, 0, 1);
      __builtin_prefetch(V + (size_t)(32 + lane) * S_ + k0 + 64, 0, 1);
    }

    float p[4][8];
#pragma unroll
    for (int r = 0; r < 8; ++r) {
      const int qr = qrow_base + r;
      const unsigned char* mr = M + (size_t)qr * S_ + k0 + (lane & 15);
      float v[4];
      float mx = -3.0e38f;
#pragma unroll
      for (int tt = 0; tt < 4; ++tt) {
        const bool mv = mr[16 * tt] != 0;
        v[tt] = mv ? s[tt][r] * scale : -3.0e38f;
        mx = fmaxf(mx, v[tt]);
      }
#pragma unroll
      for (int off = 1; off < 16; off <<= 1) mx = fmaxf(mx, __shfl_xor(mx, off, 32));
      const float mnew = fmaxf(mrun[r], mx);
      float psum = 0.0f;
#pragma unroll
      for (int tt = 0; tt < 4; ++tt) {
        const float ev = (v[tt] > -1.0e37f) ? __expf(v[tt] - mnew) : 0.0f;
        p[tt][r] = ev;
        psum += ev;
      }
#pragma unroll
      for (int off = 1; off < 16; off <<= 1) psum += __shfl_xor(psum, off, 32);
      const float alpha = __expf(mrun[r] - mnew);
      mrun[r] = mnew;
      lrun[r] = lrun[r] * alpha + psum;
      acc0[r] *= alpha; acc1[r] *= alpha; acc2[r] *= alpha; acc3[r] *= alpha;
    }

    // C-layout -> A-layout re-shuffle of P through LDS
    __syncthreads();
    const int prow = (lane >> 4) << 3;
#pragma unroll
    for (int r = 0; r < 8; ++r) {
#pragma unroll
      for (int tt = 0; tt < 4; ++tt)
        plds[wv][prow + r][16 * tt + (lane & 15)] = (bf16)p[tt][r];
    }
    __syncthreads();
    v16bf pa0, pa1;
    {
      const bf16* pl = &plds[wv][lane & 15][(lane >> 4) << 3];
#pragma unroll
      for (int i = 0; i < 8; ++i) {
        pa0[i] = pl[i];      pa0[i + 8] = pl[i + 16];
        pa1[i] = pl[i + 32]; pa1[i + 8] = pl[i + 48];
      }
    }

    // P @ V  (A = P 16x64 in two frags, B = V chunk, 4 d-tiles of 16)
    acc0 = wmma_bf16(pa0, load_bT_frag(V, S_, 0,  k0,      lane), acc0);
    acc1 = wmma_bf16(pa0, load_bT_frag(V, S_, 16, k0,      lane), acc1);
    acc2 = wmma_bf16(pa0, load_bT_frag(V, S_, 32, k0,      lane), acc2);
    acc3 = wmma_bf16(pa0, load_bT_frag(V, S_, 48, k0,      lane), acc3);
    acc0 = wmma_bf16(pa1, load_bT_frag(V, S_, 0,  k0 + 32, lane), acc0);
    acc1 = wmma_bf16(pa1, load_bT_frag(V, S_, 16, k0 + 32, lane), acc1);
    acc2 = wmma_bf16(pa1, load_bT_frag(V, S_, 32, k0 + 32, lane), acc2);
    acc3 = wmma_bf16(pa1, load_bT_frag(V, S_, 48, k0 + 32, lane), acc3);
  }

  // normalize and scatter into [b,s, (d*16 + h)] bf16 layout for the out-proj GEMM
  const int n = lane & 15;
#pragma unroll
  for (int r = 0; r < 8; ++r) {
    const float inv = lrun[r] > 0.0f ? 1.0f / lrun[r] : 0.0f;
    const int qr = qrow_base + r;
    bf16* dst = atn + ((size_t)(bb * S_ + qr)) * D_ + h;
    dst[(size_t)((0  + n) << 4)] = (bf16)(acc0[r] * inv);
    dst[(size_t)((16 + n) << 4)] = (bf16)(acc1[r] * inv);
    dst[(size_t)((32 + n) << 4)] = (bf16)(acc2[r] * inv);
    dst[(size_t)((48 + n) << 4)] = (bf16)(acc3[r] * inv);
  }
}

// ---------------------------------------------------------------- out GEMM
// 32x64 tile per wave, f32 output straight to d_out.
__global__ __launch_bounds__(128) void out_gemm(const bf16* __restrict__ A,
                                                const bf16* __restrict__ W,
                                                float* __restrict__ out) {
  const int lane = threadIdx.x & 31;
  const int wv   = threadIdx.x >> 5;
  const int row0 = blockIdx.x * 32;                 // 0..4064
  const int col0 = (blockIdx.y * 4 + wv) * 64;      // 0..960
  const v8f vzero = {0.f, 0.f, 0.f, 0.f, 0.f, 0.f, 0.f, 0.f};
  v8f acc[8];
#pragma unroll
  for (int i = 0; i < 8; ++i) acc[i] = vzero;

  for (int kc = 0; kc < D_; kc += 32) {
    v16bf a0 = load_a_frag(A, D_, row0, kc, lane);
    v16bf a1 = load_a_frag(A, D_, row0 + 16, kc, lane);
#pragma unroll
    for (int g = 0; g < 4; ++g) {
      v16bf b = load_bT_frag(W, D_, col0 + g * 16, kc, lane);
      acc[2 * g]     = wmma_bf16(a0, b, acc[2 * g]);
      acc[2 * g + 1] = wmma_bf16(a1, b, acc[2 * g + 1]);
    }
    if (kc + 128 < D_) {
      __builtin_prefetch(A + (size_t)(row0 + lane) * D_ + kc + 128, 0, 1);
      __builtin_prefetch(W + (size_t)(col0 + (lane & 31)) * D_ + kc + 128, 0, 1);
      __builtin_prefetch(W + (size_t)(col0 + 32 + (lane & 31)) * D_ + kc + 128, 0, 1);
    }
  }

#pragma unroll
  for (int g = 0; g < 4; ++g) {
    const int col = col0 + g * 16 + (lane & 15);
#pragma unroll
    for (int rblk = 0; rblk < 2; ++rblk) {
      const int rb = row0 + rblk * 16 + ((lane >> 4) << 3);
      float* dst = out + (size_t)rb * D_ + col;
      const v8f a = acc[2 * g + rblk];
#pragma unroll
      for (int r = 0; r < 8; ++r) dst[(size_t)r * D_] = a[r];
    }
  }
}

// ---------------------------------------------------------------- launcher
extern "C" void kernel_launch(void* const* d_in, const int* in_sizes, int n_in,
                              void* d_out, int out_size, void* d_ws, size_t ws_size,
                              hipStream_t stream) {
  const float* xs    = (const float*)d_in[0];
  const unsigned char* mask = (const unsigned char*)d_in[1];
  const float* w_qkv = (const float*)d_in[2];
  const float* w_out = (const float*)d_in[3];
  float* out = (float*)d_out;

  char* ws = (char*)d_ws;
  const size_t n_xs   = (size_t)B_ * S_ * D_;        // 4,194,304
  const size_t n_wqkv = (size_t)3 * D_ * D_;         // 3,145,728
  const size_t n_wout = (size_t)D_ * D_;             // 1,048,576
  bf16* xs_bf   = (bf16*)(ws);                                   // 8 MiB
  bf16* wqkv_bf = (bf16*)(ws + 8u  * 1024 * 1024);               // 6 MiB
  bf16* wout_bf = (bf16*)(ws + 14u * 1024 * 1024);               // 2 MiB
  bf16* Qb      = (bf16*)(ws + 16u * 1024 * 1024);               // 8 MiB
  bf16* Kb      = (bf16*)(ws + 24u * 1024 * 1024);               // 8 MiB
  bf16* Vt      = (bf16*)(ws + 32u * 1024 * 1024);               // 8 MiB
  bf16* atn     = (bf16*)(ws + 40u * 1024 * 1024);               // 8 MiB

  cvt_f32_bf16<<<(int)((n_xs + 255) / 256),   256, 0, stream>>>(xs, xs_bf, (int)n_xs);
  cvt_f32_bf16<<<(int)((n_wqkv + 255) / 256), 256, 0, stream>>>(w_qkv, wqkv_bf, (int)n_wqkv);
  cvt_f32_bf16<<<(int)((n_wout + 255) / 256), 256, 0, stream>>>(w_out, wout_bf, (int)n_wout);

  qkv_gemm<<<dim3(128, 12), 128, 0, stream>>>(xs_bf, wqkv_bf, Qb, Kb, Vt);

  attn_kernel<<<512, 256, 0, stream>>>(Qb, Kb, Vt, mask, atn);

  out_gemm<<<dim3(128, 4), 128, 0, stream>>>(atn, wout_bf, out);
}
